// FirstStage_87875030876869
// MI455X (gfx1250) — matrix operand order repaired and checked by reading, not compiled, not run on hardware
//
#include <hip/hip_runtime.h>
#include <hip/hip_bf16.h>

// ---------------- problem constants ----------------
constexpr int kB = 64, kN = 256, kC = 384, kH = 8, kHD = 48, kDEPTH = 6;
constexpr int kHDP = 64;                 // head dim padded to 64 for K-chunks of 32
constexpr int kM = kB * kN;              // 16384 token rows
constexpr int kBH = kB * kH;             // 512 (batch, head) pairs
constexpr int k3C = 3 * kC;              // 1152
#define LN_EPS 1e-5f

typedef _Float16 half_t;
typedef __attribute__((ext_vector_type(16))) _Float16 v16h;
typedef __attribute__((ext_vector_type(8)))  _Float16 v8h;
typedef __attribute__((ext_vector_type(8)))  float    v8f;

// ---------------- WMMA helpers ----------------
__device__ __forceinline__ v8f wmma_f16(v16h a, v16h b, v8f c) {
  // (neg_a, A, neg_b, B, c_mod, C, reuse_a, reuse_b)
  return __builtin_amdgcn_wmma_f32_16x16x32_f16(false, a, false, b, (short)0, c,
                                                false, false);
}

// A-fragment (16x32, 16-bit): lanes 0-15 hold M=lane, K={0..7,16..23};
// lanes 16-31 hold M=lane-16, K={8..15,24..31}.  p = subtile origin, ld = row stride.
__device__ __forceinline__ v16h frag_a(const half_t* p, int ld, int lane) {
  int row = lane & 15;
  int kb = (lane < 16) ? 0 : 8;
  const half_t* q = p + (size_t)row * ld + kb;
  v16h x;
  ((v8h*)&x)[0] = *(const v8h*)(q);        // K kb..kb+7
  ((v8h*)&x)[1] = *(const v8h*)(q + 16);   // K kb+16..kb+23
  return x;
}

// B-fragment (32x16, 16-bit): lanes 0-15 hold N=lane, K={0..7,16..23};
// lanes 16-31 hold N=lane-16, K={8..15,24..31}.  p = subtile origin (row = N, cols = K), ld = row stride.
__device__ __forceinline__ v16h frag_b(const half_t* p, int ld, int lane) {
  int col = lane & 15;
  int kb = (lane < 16) ? 0 : 8;
  const half_t* q = p + (size_t)col * ld + kb;
  v16h x;
  ((v8h*)&x)[0] = *(const v8h*)(q);
  ((v8h*)&x)[1] = *(const v8h*)(q + 16);
  return x;
}

// ---------------- async global -> LDS copy (CDNA5, ASYNCcnt-tracked) ----------------
// CDNA5 ISA: each lane copies 16B from gaddr to LDS byte offset ldsByteOff.
__device__ __forceinline__ void async_copy_b128(unsigned ldsByteOff, const half_t* gaddr) {
  asm volatile("global_load_async_to_lds_b128 %0, %1, off"
               :: "v"(ldsByteOff), "v"(gaddr)
               : "memory");
}
__device__ __forceinline__ void wait_async(int n) {
  if (n == 0)
    asm volatile("s_wait_asynccnt 0x0" ::: "memory");
  else
    asm volatile("s_wait_asynccnt 0x4" ::: "memory");
}

// ---------------- fp32 -> f16 convert (weights) ----------------
__global__ void k_f2h(const float* __restrict__ in, half_t* __restrict__ out, size_t n) {
  for (size_t i = (size_t)blockIdx.x * blockDim.x + threadIdx.x; i < n;
       i += (size_t)gridDim.x * blockDim.x)
    out[i] = (half_t)in[i];
}

// ---------------- embedding stage 1: t = relu(x @ w1.T + b1) ----------------
__global__ void k_embed1(const float* __restrict__ x, const float* __restrict__ w1,
                         const float* __restrict__ b1, half_t* __restrict__ t) {
  size_t i = (size_t)blockIdx.x * blockDim.x + threadIdx.x;
  if (i >= (size_t)kM * kC) return;
  int m = (int)(i / kC), c = (int)(i % kC);
  float v = x[m * 2 + 0] * w1[c * 2 + 0] + x[m * 2 + 1] * w1[c * 2 + 1] + b1[c];
  t[i] = (half_t)fmaxf(v, 0.f);
}

// ---------------- layernorm (fp32 in -> f16 out) ----------------
__global__ void k_ln(const float* __restrict__ x, const float* __restrict__ s,
                     const float* __restrict__ b, half_t* __restrict__ y) {
  __shared__ float red[128];
  const int row = blockIdx.x, t = threadIdx.x;
  const float* xr = x + (size_t)row * kC;
  float v0 = xr[t], v1 = xr[t + 128], v2 = xr[t + 256];
  red[t] = v0 + v1 + v2;
  __syncthreads();
  for (int off = 64; off > 0; off >>= 1) {
    if (t < off) red[t] += red[t + off];
    __syncthreads();
  }
  float mu = red[0] * (1.f / kC);
  __syncthreads();
  float d0 = v0 - mu, d1 = v1 - mu, d2 = v2 - mu;
  red[t] = d0 * d0 + d1 * d1 + d2 * d2;
  __syncthreads();
  for (int off = 64; off > 0; off >>= 1) {
    if (t < off) red[t] += red[t + off];
    __syncthreads();
  }
  float rs = rsqrtf(red[0] * (1.f / kC) + LN_EPS);
  half_t* yr = y + (size_t)row * kC;
  yr[t]       = (half_t)(d0 * rs * s[t]       + b[t]);
  yr[t + 128] = (half_t)(d1 * rs * s[t + 128] + b[t + 128]);
  yr[t + 256] = (half_t)(d2 * rs * s[t + 256] + b[t + 256]);
}

// ---------------- generic WMMA GEMM: out[M,Nd] = A[M,K] @ W[Nd,K]^T (+bias)(+res) -----
// 128 threads (4 waves), 64x64 block tile, K in chunks of 32 staged into
// double-buffered dynamic LDS via global_load_async_to_lds_b128 (ASYNCcnt).
// Dynamic-LDS layout (byte offsets): A tiles at [buf*4096], W tiles at [8192 + buf*4096].
template <bool RELU, bool RES, bool WF32, bool WH>
__global__ void __launch_bounds__(128)
k_gemm(const half_t* __restrict__ A, const half_t* __restrict__ W,
       const float* __restrict__ bias, float* __restrict__ outF,
       half_t* __restrict__ outH, int Mdim, int Ndim, int Kdim) {
  extern __shared__ __align__(16) char smem[];     // 16384 bytes
  half_t* As = (half_t*)smem;                      // [2][64][32]
  half_t* Ws = (half_t*)(smem + 8192);             // [2][64][32]

  const int tid = threadIdx.x, lane = tid & 31, wave = tid >> 5;
  const int tileN = blockIdx.x * 64, tileM = blockIdx.y * 64;
  const int wm = (wave >> 1) * 32, wn = (wave & 1) * 32;

  const int lr = tid >> 1;                // loader row 0..63
  const int lhB = (tid & 1) * 32;         // loader col-half byte offset (0 or 32)
  const int lhE = (tid & 1) * 16;         // loader col-half element offset

  // stage K-chunk kk into LDS buffer `buf` (4 async ops per thread)
  auto stage = [&](int kk, int buf) {
    const half_t* gA = A + (size_t)(tileM + lr) * Kdim + kk + lhE;
    const half_t* gW = W + (size_t)(tileN + lr) * Kdim + kk + lhE;
    unsigned aOff = (unsigned)(buf * 4096 + lr * 64 + lhB);
    unsigned wOff = (unsigned)(8192 + buf * 4096 + lr * 64 + lhB);
    async_copy_b128(aOff, gA);
    async_copy_b128(aOff + 16, gA + 8);
    async_copy_b128(wOff, gW);
    async_copy_b128(wOff + 16, gW + 8);
  };

  v8f acc[2][2] = {};
  const int nk = Kdim / 32;

  stage(0, 0);
  for (int t = 0; t < nk; ++t) {
    const int buf = t & 1;
    if (t + 1 < nk) stage((t + 1) * 32, (t + 1) & 1);
    // in-order async completion: leaving 4 outstanding means tile t has landed
    wait_async((t + 1 < nk) ? 4 : 0);
    __syncthreads();

    half_t* At = As + buf * 2048;   // elements
    half_t* Wt = Ws + buf * 2048;
    v16h a0 = frag_a(At + (size_t)wm * 32, 32, lane);
    v16h a1 = frag_a(At + (size_t)(wm + 16) * 32, 32, lane);
    v16h b0 = frag_b(Wt + (size_t)wn * 32, 32, lane);
    v16h b1 = frag_b(Wt + (size_t)(wn + 16) * 32, 32, lane);
    acc[0][0] = wmma_f16(a0, b0, acc[0][0]);
    acc[0][1] = wmma_f16(a0, b1, acc[0][1]);
    acc[1][0] = wmma_f16(a1, b0, acc[1][0]);
    acc[1][1] = wmma_f16(a1, b1, acc[1][1]);
    __syncthreads();  // all waves done reading `buf` before it is re-staged
  }

  for (int i = 0; i < 2; ++i)
    for (int j = 0; j < 2; ++j) {
      int gcol = tileN + wn + j * 16 + (lane & 15);
      float bv = bias ? bias[gcol] : 0.f;
      for (int v = 0; v < 8; ++v) {
        int rowT = wm + i * 16 + ((lane < 16) ? v : v + 8);
        size_t idx = (size_t)(tileM + rowT) * Ndim + gcol;
        float val = acc[i][j][v] + bv;
        if (RELU) val = fmaxf(val, 0.f);
        if (RES) val += outF[idx];
        if (WF32 || RES) outF[idx] = val;
        if (WH) outH[idx] = (half_t)val;
      }
    }
}

// ---------------- RoPE + repack q/k/v into per-head padded layouts ----------------
// qkv (f16, [M][3C]) -> qh/kh: [BH][N][64] (d-padded, q scaled), vt: [BH][64][N] (transposed)
__global__ void k_rope(const half_t* __restrict__ qkv, half_t* __restrict__ qh,
                       half_t* __restrict__ kh, half_t* __restrict__ vt) {
  int idx = blockIdx.x * blockDim.x + threadIdx.x;
  if (idx >= kBH * kN) return;
  int bh = idx / kN, n = idx % kN;
  int bb = bh / kH, hh = bh % kH;
  const half_t* src = qkv + (size_t)(bb * kN + n) * k3C;
  const half_t* qs = src + hh * kHD;
  const half_t* ks = src + kC + hh * kHD;
  const half_t* vs = src + 2 * kC + hh * kHD;
  half_t* qd = qh + ((size_t)bh * kN + n) * kHDP;
  half_t* kd = kh + ((size_t)bh * kN + n) * kHDP;
  const float scale = rsqrtf((float)kHD);
  const float pos = (float)n;
  for (int d = 0; d < kHD / 2; ++d) {
    float inv = __powf(10000.f, -(2.f * d) / (float)kHD);
    float f = pos * inv, sn, cs;
    __sincosf(f, &sn, &cs);
    float qu = (float)qs[2 * d], qe = (float)qs[2 * d + 1];
    qd[2 * d]     = (half_t)((qu * cs - qe * sn) * scale);
    qd[2 * d + 1] = (half_t)((qu * sn + qe * cs) * scale);
    float ku = (float)ks[2 * d], ke = (float)ks[2 * d + 1];
    kd[2 * d]     = (half_t)(ku * cs - ke * sn);
    kd[2 * d + 1] = (half_t)(ku * sn + ke * cs);
  }
  for (int d = kHD; d < kHDP; ++d) { qd[d] = (half_t)0.f; kd[d] = (half_t)0.f; }
  half_t* vcol = vt + (size_t)bh * kHDP * kN + n;
  for (int d = 0; d < kHD; ++d)  vcol[(size_t)d * kN] = vs[d];
  for (int d = kHD; d < kHDP; ++d) vcol[(size_t)d * kN] = (half_t)0.f;
}

// ---------------- attention: one wave handles (b,h) x 16 query rows ----------------
__global__ void __launch_bounds__(32)
k_attn(const half_t* __restrict__ qh, const half_t* __restrict__ kh,
       const half_t* __restrict__ vt, const float* __restrict__ mask,
       half_t* __restrict__ o) {
  __shared__ __align__(16) float  sc[16][kN];   // 16 KB raw scores
  __shared__ __align__(16) half_t ps[16][kN];   //  8 KB softmax probs (f16)
  const int bh = blockIdx.y, bb = bh / kH, hh = bh % kH;
  const int q0 = blockIdx.x * 16;
  const int lane = threadIdx.x;

  const half_t* qbase = qh + ((size_t)bh * kN + q0) * kHDP;
  const half_t* kbase = kh + (size_t)bh * kN * kHDP;

  // S = (q*scale) @ k^T  (K-dim = head dim, padded to 64 -> two K=32 chunks)
  for (int kt = 0; kt < kN / 16; ++kt) {
    v8f accS = {};
    for (int kc = 0; kc < kHDP; kc += 32) {
      v16h a = frag_a(qbase + kc, kHDP, lane);
      v16h b = frag_b(kbase + (size_t)(kt * 16) * kHDP + kc, kHDP, lane);
      accS = wmma_f16(a, b, accS);
    }
    int col = kt * 16 + (lane & 15);
    float mkey = mask[bb * kN + col];
    for (int v = 0; v < 8; ++v) {
      int r = (lane < 16) ? v : v + 8;
      sc[r][col] = accS[v] + mkey + mask[bb * kN + q0 + r];
    }
  }
  __syncthreads();

  // row-wise softmax over 256 keys (8 elems/lane, wave32 shuffle reductions)
  for (int r = 0; r < 16; ++r) {
    float vals[8], mx = -1e30f;
    for (int i = 0; i < 8; ++i) {
      vals[i] = sc[r][lane + i * 32];
      mx = fmaxf(mx, vals[i]);
    }
    for (int off = 16; off > 0; off >>= 1) mx = fmaxf(mx, __shfl_xor(mx, off, 32));
    float sum = 0.f;
    for (int i = 0; i < 8; ++i) { vals[i] = __expf(vals[i] - mx); sum += vals[i]; }
    for (int off = 16; off > 0; off >>= 1) sum += __shfl_xor(sum, off, 32);
    float inv = 1.f / sum;
    for (int i = 0; i < 8; ++i) ps[r][lane + i * 32] = (half_t)(vals[i] * inv);
  }
  __syncthreads();

  // O = P @ V   (V transposed: vt[bh][d][key], so B-fragment loads are contiguous)
  const half_t* vbase = vt + (size_t)bh * kHDP * kN;
  for (int nt = 0; nt < 3; ++nt) {  // 3 * 16 = 48 output dims
    v8f accO = {};
    for (int kt = 0; kt < kN; kt += 32) {
      v16h a = frag_a(&ps[0][kt], kN, lane);
      v16h b = frag_b(vbase + (size_t)(nt * 16) * kN + kt, kN, lane);
      accO = wmma_f16(a, b, accO);
    }
    int d = nt * 16 + (lane & 15);
    for (int v = 0; v < 8; ++v) {
      int r = (lane < 16) ? v : v + 8;
      o[((size_t)(bb * kN) + q0 + r) * kC + hh * kHD + d] = (half_t)accO[v];
    }
  }
}

// ---------------- host side ----------------
extern "C" void kernel_launch(void* const* d_in, const int* in_sizes, int n_in,
                              void* d_out, int out_size, void* d_ws, size_t ws_size,
                              hipStream_t stream) {
  const float* x      = (const float*)d_in[0];
  const float* mask   = (const float*)d_in[1];
  const float* emb_w1 = (const float*)d_in[2];
  const float* emb_b1 = (const float*)d_in[3];
  const float* emb_w2 = (const float*)d_in[4];
  const float* emb_b2 = (const float*)d_in[5];
  const float* qkv_w  = (const float*)d_in[6];
  const float* proj_w = (const float*)d_in[7];
  const float* proj_b = (const float*)d_in[8];
  const float* mlp_w1 = (const float*)d_in[9];
  const float* mlp_b1 = (const float*)d_in[10];
  const float* mlp_w2 = (const float*)d_in[11];
  const float* mlp_b2 = (const float*)d_in[12];
  const float* ln1_s  = (const float*)d_in[13];
  const float* ln1_b  = (const float*)d_in[14];
  const float* ln2_s  = (const float*)d_in[15];
  const float* ln2_b  = (const float*)d_in[16];

  // workspace carve-out (256B aligned chunks)
  char* wp = (char*)d_ws;
  auto carve = [&](size_t bytes) -> void* {
    void* p = wp;
    wp += (bytes + 255) & ~(size_t)255;
    return p;
  };
  half_t* wEmb2 = (half_t*)carve((size_t)kC * kC * 2);
  half_t* wQkv  = (half_t*)carve((size_t)kDEPTH * k3C * kC * 2);
  half_t* wProj = (half_t*)carve((size_t)kDEPTH * kC * kC * 2);
  half_t* wMlp1 = (half_t*)carve((size_t)kDEPTH * kC * kC * 2);
  half_t* wMlp2 = (half_t*)carve((size_t)kDEPTH * kC * kC * 2);
  float*  hbuf  = (float*) carve((size_t)kM * kC * 4);      // fp32 residual stream
  half_t* ybuf  = (half_t*)carve((size_t)kM * kC * 2);      // LN / embed-t output (f16)
  half_t* qkvb  = (half_t*)carve((size_t)kM * k3C * 2);     // qkv projection (f16)
  half_t* qhb   = (half_t*)carve((size_t)kBH * kN * kHDP * 2);
  half_t* khb   = (half_t*)carve((size_t)kBH * kN * kHDP * 2);
  half_t* vtb   = (half_t*)carve((size_t)kBH * kHDP * kN * 2);
  half_t* obuf  = (half_t*)carve((size_t)kM * kC * 2);
  half_t* m1buf = (half_t*)carve((size_t)kM * kC * 2);

  constexpr size_t kGemmLds = 16384;  // 2 x (A 4KB + W 4KB)

  // convert all weights fp32 -> f16 once
  k_f2h<<<2048, 256, 0, stream>>>(emb_w2, wEmb2, (size_t)kC * kC);
  k_f2h<<<4096, 256, 0, stream>>>(qkv_w, wQkv, (size_t)kDEPTH * k3C * kC);
  k_f2h<<<2048, 256, 0, stream>>>(proj_w, wProj, (size_t)kDEPTH * kC * kC);
  k_f2h<<<2048, 256, 0, stream>>>(mlp_w1, wMlp1, (size_t)kDEPTH * kC * kC);
  k_f2h<<<2048, 256, 0, stream>>>(mlp_w2, wMlp2, (size_t)kDEPTH * kC * kC);

  // embedding
  k_embed1<<<(kM * kC + 255) / 256, 256, 0, stream>>>(x, emb_w1, emb_b1, ybuf);
  k_gemm<false, false, true, false>
      <<<dim3(kC / 64, kM / 64), 128, kGemmLds, stream>>>(
          ybuf, wEmb2, emb_b2, hbuf, nullptr, kM, kC, kC);

  for (int i = 0; i < kDEPTH; ++i) {
    k_ln<<<kM, 128, 0, stream>>>(hbuf, ln1_s + i * kC, ln1_b + i * kC, ybuf);
    k_gemm<false, false, false, true>
        <<<dim3(k3C / 64, kM / 64), 128, kGemmLds, stream>>>(
            ybuf, wQkv + (size_t)i * k3C * kC, nullptr, nullptr, qkvb, kM, k3C, kC);
    k_rope<<<(kBH * kN + 127) / 128, 128, 0, stream>>>(qkvb, qhb, khb, vtb);
    k_attn<<<dim3(kN / 16, kBH), 32, 0, stream>>>(qhb, khb, vtb, mask, obuf);
    k_gemm<false, true, true, false>
        <<<dim3(kC / 64, kM / 64), 128, kGemmLds, stream>>>(
            obuf, wProj + (size_t)i * kC * kC, proj_b + i * kC, hbuf, nullptr, kM, kC, kC);
    k_ln<<<kM, 128, 0, stream>>>(hbuf, ln2_s + i * kC, ln2_b + i * kC, ybuf);
    k_gemm<true, false, false, true>
        <<<dim3(kC / 64, kM / 64), 128, kGemmLds, stream>>>(
            ybuf, wMlp1 + (size_t)i * kC * kC, mlp_b1 + i * kC, nullptr, m1buf, kM, kC, kC);
    k_gemm<false, true, true, false>
        <<<dim3(kC / 64, kM / 64), 128, kGemmLds, stream>>>(
            m1buf, wMlp2 + (size_t)i * kC * kC, mlp_b2 + i * kC, hbuf, nullptr, kM, kC, kC);
  }

  hipMemcpyAsync(d_out, hbuf, (size_t)kM * kC * sizeof(float),
                 hipMemcpyDeviceToDevice, stream);
}
